// Net_5188320494138
// MI455X (gfx1250) — matrix operand order repaired and checked by reading, not compiled, not run on hardware
//
#include <hip/hip_runtime.h>
#include <hip/hip_bf16.h>

typedef __attribute__((ext_vector_type(16))) _Float16 v16h;
typedef __attribute__((ext_vector_type(8)))  float    v8f;
typedef __attribute__((ext_vector_type(4)))  unsigned int u32x4;
typedef __attribute__((ext_vector_type(8)))  int      i32x8;
typedef __attribute__((ext_vector_type(4)))  int      i32x4;

// ---------------------------------------------------------------------------
// WMMA fragment helpers (16x16x32 f16, wave32; A/B use contiguous k, not AMD's table order)
// A (16x32, M=lane&15):  element j -> K = hi*16 + j
// B (32x16, N=lane&15):  element j -> K = hi*16 + j
// C (16x16): VGPR r -> M = hi*8 + r, N = lane&15
// ---------------------------------------------------------------------------
__device__ __forceinline__ v16h frag_a_lds(const _Float16* base, int stride,
                                           int m, int hi, int kBase) {
  v16h a;
#pragma unroll
  for (int j = 0; j < 16; ++j) {
    int k = hi * 16 + j;
    a[j] = base[m * stride + kBase + k];
  }
  return a;
}

__device__ __forceinline__ v16h frag_b_lds(const _Float16* base, int stride,
                                           int n, int hi, int kBase) {
  v16h b;
#pragma unroll
  for (int j = 0; j < 16; ++j)
    b[j] = base[(kBase + hi * 16 + j) * stride + n];
  return b;
}

// ---------------------------------------------------------------------------
// Stage 1: NeRF MLP.  Each wave processes a 16-ray tile; weights staged in LDS
// as f16; activations in per-wave LDS tile [h(0..63) | rays(64..162) | pad].
// ---------------------------------------------------------------------------
__device__ __forceinline__ void mlp_gemm_layer(const _Float16* w, const float* bias,
                                               _Float16* act, int aColBase,
                                               int kTiles, int ln, int hi) {
  v8f cc[4];
#pragma unroll
  for (int nt = 0; nt < 4; ++nt)
#pragma unroll
    for (int e = 0; e < 8; ++e) cc[nt][e] = 0.f;

  for (int t = 0; t < kTiles; ++t) {
    v16h a = frag_a_lds(act, 192, ln, hi, aColBase + 32 * t);
#pragma unroll
    for (int nt = 0; nt < 4; ++nt) {
      v16h b = frag_b_lds(w, 64, nt * 16 + ln, hi, 32 * t);
      cc[nt] = __builtin_amdgcn_wmma_f32_16x16x32_f16(
          false, a, false, b, (short)0, cc[nt], false, false);
    }
  }
#pragma unroll
  for (int nt = 0; nt < 4; ++nt) {
    float bv = bias[nt * 16 + ln];
#pragma unroll
    for (int r = 0; r < 8; ++r) {
      int m = hi * 8 + r;
      float v = cc[nt][r] + bv;
      v = fmaxf(v, 0.f);                 // relu
      act[m * 192 + nt * 16 + ln] = (_Float16)v;
    }
  }
}

__global__ __launch_bounds__(64) void mlp_kernel(
    const float* __restrict__ rays,
    const float* __restrict__ W_in, const float* __restrict__ b_in,
    const float* __restrict__ W_dense, const float* __restrict__ b_dense,
    const float* __restrict__ W_reg, const float* __restrict__ b_reg,
    const float* __restrict__ W_out, const float* __restrict__ b_out,
    float* __restrict__ mlp_out) {
  __shared__ _Float16 sWin[128 * 64];    // 99 -> padded 128 rows
  __shared__ _Float16 sWd[64 * 64];
  __shared__ _Float16 sWreg[192 * 64];   // 163 -> padded 192 rows
  __shared__ _Float16 sWout[64 * 16];    // 4 -> padded 16 cols
  __shared__ _Float16 sAct[2][16 * 192]; // per-wave activation tile

  const int tid = threadIdx.x;
  const int wave = tid >> 5;
  const int lane = tid & 31;
  const int ln = lane & 15;
  const int hi = lane >> 4;

  for (int i = tid; i < 128 * 64; i += 64) {
    int r = i >> 6;
    sWin[i] = (_Float16)(r < 99 ? W_in[i] : 0.f);
  }
  for (int i = tid; i < 64 * 64; i += 64) sWd[i] = (_Float16)W_dense[i];
  for (int i = tid; i < 192 * 64; i += 64) {
    int r = i >> 6;
    sWreg[i] = (_Float16)(r < 163 ? W_reg[i] : 0.f);
  }
  for (int i = tid; i < 64 * 16; i += 64) {
    int r = i >> 4, c = i & 15;
    sWout[i] = (_Float16)(c < 4 ? W_out[r * 4 + c] : 0.f);
  }

  _Float16* act = sAct[wave];
  const long rowBase = (long)blockIdx.x * 32 + wave * 16;
  for (int i = lane; i < 16 * 128; i += 32) {
    int r = i >> 7, c = i & 127;
    float v = (c < 99) ? rays[(rowBase + r) * 99 + c] : 0.f;
    act[r * 192 + 64 + c] = (_Float16)v;   // rays at cols 64..162, pad zeros
  }
  __syncthreads();

  // input layer: relu(rays @ W_in + b_in), K = 128 (padded)
  mlp_gemm_layer(sWin, b_in, act, 64, 4, ln, hi);
  // 8 shared dense layers, reg (skip-concat) layer after i==4
  for (int i = 0; i < 8; ++i) {
    mlp_gemm_layer(sWd, b_dense, act, 0, 2, ln, hi);
    if (i == 4)
      mlp_gemm_layer(sWreg, b_reg, act, 0, 6, ln, hi);  // [h|rays] K=192
  }
  // output layer: h @ W_out + b_out (no relu)
  v8f co;
#pragma unroll
  for (int e = 0; e < 8; ++e) co[e] = 0.f;
  for (int t = 0; t < 2; ++t) {
    v16h a = frag_a_lds(act, 192, ln, hi, 32 * t);
    v16h b = frag_b_lds(sWout, 16, ln, hi, 32 * t);
    co = __builtin_amdgcn_wmma_f32_16x16x32_f16(
        false, a, false, b, (short)0, co, false, false);
  }
  if (ln < 4) {
    float bv = b_out[ln];
#pragma unroll
    for (int r = 0; r < 8; ++r) {
      int m = hi * 8 + r;
      mlp_out[(rowBase + m) * 4 + ln] = co[r] + bv;
    }
  }
}

// ---------------------------------------------------------------------------
// Stage 2: volume rendering.  One thread per pixel; rgb written already
// transposed to [B,3,W,H] for the CNN, depth to tail of d_out.
// ---------------------------------------------------------------------------
__global__ void render_kernel(const float* __restrict__ mlp_out,
                              const float* __restrict__ t_vals,
                              float* __restrict__ rgb_out,   // [2,3,64,64] (W,H)
                              float* __restrict__ depth_out) {
  int pix = blockIdx.x * blockDim.x + threadIdx.x;
  if (pix >= 2 * 64 * 64) return;
  int b = pix >> 12;
  int rem = pix & 4095;
  int h = rem >> 6;
  int w = rem & 63;

  const float* o = mlp_out + (size_t)pix * 64 * 4;
  const float* tv = t_vals + (size_t)pix * 64;
  float T = 1.f, r = 0.f, g = 0.f, bl = 0.f, d = 0.f;
  for (int s = 0; s < 64; ++s) {
    float sr = 1.f / (1.f + __expf(-o[s * 4 + 0]));
    float sg = 1.f / (1.f + __expf(-o[s * 4 + 1]));
    float sb = 1.f / (1.f + __expf(-o[s * 4 + 2]));
    float sigma = fmaxf(o[s * 4 + 3], 0.f);
    float delta = (s < 63) ? (tv[s + 1] - tv[s]) : 1e10f;
    float alpha = 1.f - __expf(-sigma * delta);
    T *= (1.f - alpha + 1e-10f);          // inclusive cumprod
    float wt = alpha * T;
    r += wt * sr; g += wt * sg; bl += wt * sb;
    d += wt * tv[s];
  }
  rgb_out[((b * 3 + 0) * 64 + w) * 64 + h] = r;
  rgb_out[((b * 3 + 1) * 64 + w) * 64 + h] = g;
  rgb_out[((b * 3 + 2) * 64 + w) * 64 + h] = bl;
  depth_out[pix] = d;
}

// ---------------------------------------------------------------------------
// Stage 3: implicit-GEMM conv via WMMA, LDS-tiled.
// Block = 128 threads (4 waves): 64 pixels x 64 out-channels.
// Per 32-wide K chunk:
//   - B tile (64 co x 32 k, f32) loaded by the Tensor Data Mover (TDM) when
//     available (D# built per CDNA5 ISA), else manually.
//     tensor_dim = remaining K/Cout so TDM OOB zero-fill handles edge tiles.
//   - A tile (64 px x 32 k, f16) im2col-gathered once, shared by all 4 waves.
//   - each wave: 1 B fragment x 4 pixel-subtile WMMAs.
// Fused bias + optional residual-add + optional PReLU epilogue.
// ---------------------------------------------------------------------------
#if __has_builtin(__builtin_amdgcn_tensor_load_to_lds)
#define CONV_USE_TDM 1
#else
#define CONV_USE_TDM 0
#endif

__global__ __launch_bounds__(128) void conv_wmma_tiled(
    const float* __restrict__ in, const float* __restrict__ wgt,
    const float* __restrict__ bias, const float* __restrict__ resid,
    const float* __restrict__ prelu, float* __restrict__ out,
    int Cin, int Cout, int Hh, int Ww, int Ksz) {
  __shared__ _Float16 sA[64 * 32];   // [m][k], 4 KB
  __shared__ float    sB[64 * 32];   // [co][k], 8 KB (TDM destination)

  const int tid = threadIdx.x;
  const int wave = tid >> 5;
  const int lane = tid & 31;
  const int ln = lane & 15;
  const int hi = lane >> 4;
  const int pixBase = blockIdx.x * 64;
  const int coBase = blockIdx.y * 64;
  const int bidx = blockIdx.z;
  const int npix = Hh * Ww;
  const int Ks2 = Ksz * Ksz;
  const int Kdim = Cin * Ks2;
  const int pad = Ksz >> 1;
  const float* inB = in + (size_t)bidx * Cin * npix;

  v8f acc[4];
#pragma unroll
  for (int st = 0; st < 4; ++st)
#pragma unroll
    for (int e = 0; e < 8; ++e) acc[st][e] = 0.f;

#if CONV_USE_TDM
  const unsigned ldsB =
      (unsigned)(unsigned long long)(__attribute__((address_space(3))) float*)sB;
#endif

  for (int kBase = 0; kBase < Kdim; kBase += 32) {
#if CONV_USE_TDM
    // ---- B tile via Tensor Data Mover (wave 0 issues the DMA) ----
    if (wave == 0) {
      unsigned long long ga =
          (unsigned long long)(const void*)(wgt + (size_t)coBase * Kdim + kBase);
      unsigned d0 = (unsigned)(Kdim - kBase);   // remaining K  -> OOB cols = 0
      unsigned d1 = (unsigned)(Cout - coBase);  // remaining Co -> OOB rows = 0
      unsigned long long s0 = (unsigned long long)Kdim;  // row stride (elements)
      u32x4 g0;
      g0[0] = 1u;                                        // count=1 (user D#)
      g0[1] = ldsB;                                      // lds_addr
      g0[2] = (unsigned)(ga & 0xffffffffu);              // global_addr[31:0]
      g0[3] = (unsigned)((ga >> 32) & 0x1ffffffu) | (2u << 30);  // [56:32]|type=2
      i32x8 g1;
      g1[0] = 0x20000;                                   // data_size=2 (4 bytes)
      g1[1] = (int)((d0 & 0xffffu) << 16);               // dim0[15:0] @63:48
      g1[2] = (int)(((d0 >> 16) & 0xffffu) | ((d1 & 0xffffu) << 16));
      g1[3] = (int)(((d1 >> 16) & 0xffffu) | (32u << 16));   // tile_dim0=32
      g1[4] = 64;                                        // tile_dim1=64, tile_dim2=0
      g1[5] = (int)(s0 & 0xffffffffu);                   // dim0_stride[31:0]
      g1[6] = (int)((s0 >> 32) & 0xffffu);               // dim0_stride[47:32]
      g1[7] = 0;
      i32x4 z4;
      z4[0] = 0; z4[1] = 0; z4[2] = 0; z4[3] = 0;
      i32x8 z8;
#pragma unroll
      for (int e = 0; e < 8; ++e) z8[e] = 0;
      __builtin_amdgcn_tensor_load_to_lds(g0, g1, z4, z4, z8, 0);
#if __has_builtin(__builtin_amdgcn_s_wait_tensorcnt)
      __builtin_amdgcn_s_wait_tensorcnt(0);
#else
      asm volatile("s_wait_tensorcnt 0x0" ::: "memory");
#endif
    }
#else
    // ---- B tile: manual staging ----
    for (int i = tid; i < 64 * 32; i += 128) {
      int co = i >> 5, k = i & 31;
      int kg = kBase + k;
      float v = 0.f;
      if (kg < Kdim && coBase + co < Cout)
        v = wgt[(size_t)(coBase + co) * Kdim + kg];
      sB[i] = v;
    }
#endif
    // ---- A tile: im2col gather (shared across the 4 waves) ----
    for (int i = tid; i < 64 * 32; i += 128) {
      int m = i >> 5, k = i & 31;
      int kg = kBase + k;
      int p = pixBase + m;
      float v = 0.f;
      if (kg < Kdim && p < npix) {
        int c = kg / Ks2;
        int remk = kg - c * Ks2;
        int ky = remk / Ksz, kx = remk - ky * Ksz;
        int oy = p / Ww, ox = p - oy * Ww;
        int iy = oy + ky - pad, ix = ox + kx - pad;
        if ((unsigned)iy < (unsigned)Hh && (unsigned)ix < (unsigned)Ww)
          v = inB[(size_t)c * npix + iy * Ww + ix];
      }
      sA[i] = (_Float16)v;
    }
    __syncthreads();

    // ---- compute: one B fragment per wave, 4 pixel-subtile WMMAs ----
    v16h bf;
#pragma unroll
    for (int j = 0; j < 16; ++j)
      bf[j] = (_Float16)sB[(wave * 16 + ln) * 32 + hi * 16 + j];
#pragma unroll
    for (int st = 0; st < 4; ++st) {
      v16h a = frag_a_lds(sA + st * 16 * 32, 32, ln, hi, 0);
      acc[st] = __builtin_amdgcn_wmma_f32_16x16x32_f16(
          false, a, false, bf, (short)0, acc[st], false, false);
    }
    __syncthreads();   // protect tiles before next chunk restage
  }

  // ---- epilogue: bias (+residual) (+PReLU) ----
  int co = coBase + wave * 16 + ln;
  bool cok = co < Cout;
  float bval = cok ? bias[co] : 0.f;
  float slope = prelu ? *prelu : 0.f;
#pragma unroll
  for (int st = 0; st < 4; ++st) {
#pragma unroll
    for (int r = 0; r < 8; ++r) {
      int p = pixBase + st * 16 + hi * 8 + r;
      if (cok && p < npix) {
        size_t oi = ((size_t)bidx * Cout + co) * npix + p;
        float v = acc[st][r] + bval;
        if (resid) v += resid[oi];
        if (prelu) v = v > 0.f ? v : slope * v;
        out[oi] = v;
      }
    }
  }
}

// ---------------------------------------------------------------------------
// pixel_shuffle(r=2) + PReLU:  in [B,4*Co,H,W] -> out [B,Co,2H,2W]
// ---------------------------------------------------------------------------
__global__ void shuffle_prelu_kernel(const float* __restrict__ in,
                                     const float* __restrict__ alpha_p,
                                     float* __restrict__ out,
                                     int Co, int Hh, int Ww) {
  const int W2 = 2 * Ww, H2 = 2 * Hh;
  size_t total = (size_t)2 * Co * H2 * W2;
  size_t i = (size_t)blockIdx.x * blockDim.x + threadIdx.x;
  if (i >= total) return;
  int x2 = (int)(i % W2);
  size_t t1 = i / W2;
  int y2 = (int)(t1 % H2);
  size_t t2 = t1 / H2;
  int c = (int)(t2 % Co);
  int b = (int)(t2 / Co);
  int r1 = y2 & 1, r2 = x2 & 1, hh = y2 >> 1, wwi = x2 >> 1;
  int ci = c * 4 + r1 * 2 + r2;
  float v = in[(((size_t)b * (4 * Co) + ci) * Hh + hh) * Ww + wwi];
  float a = *alpha_p;
  out[i] = v > 0.f ? v : a * v;
}

// ---------------------------------------------------------------------------
extern "C" void kernel_launch(void* const* d_in, const int* in_sizes, int n_in,
                              void* d_out, int out_size, void* d_ws, size_t ws_size,
                              hipStream_t stream) {
  (void)in_sizes; (void)n_in; (void)out_size; (void)ws_size;
  const float* rays      = (const float*)d_in[0];
  const float* t_vals    = (const float*)d_in[1];
  const float* W_in      = (const float*)d_in[2];
  const float* b_in      = (const float*)d_in[3];
  const float* W_dense   = (const float*)d_in[4];
  const float* b_dense   = (const float*)d_in[5];
  const float* W_reg     = (const float*)d_in[6];
  const float* b_reg     = (const float*)d_in[7];
  const float* W_out     = (const float*)d_in[8];
  const float* b_out     = (const float*)d_in[9];
  const float* g_conv1_w = (const float*)d_in[10];
  const float* g_conv1_b = (const float*)d_in[11];
  const float* g_prelu1  = (const float*)d_in[12];
  const float* res_wa    = (const float*)d_in[13];
  const float* res_ba    = (const float*)d_in[14];
  const float* res_alpha = (const float*)d_in[15];
  const float* res_wb    = (const float*)d_in[16];
  const float* res_bb    = (const float*)d_in[17];
  const float* g_mid_w   = (const float*)d_in[18];
  const float* g_mid_b   = (const float*)d_in[19];
  const float* up1_w     = (const float*)d_in[20];
  const float* up1_b     = (const float*)d_in[21];
  const float* up1_a     = (const float*)d_in[22];
  const float* up2_w     = (const float*)d_in[23];
  const float* up2_b     = (const float*)d_in[24];
  const float* up2_a     = (const float*)d_in[25];
  const float* g_out_w   = (const float*)d_in[26];
  const float* g_out_b   = (const float*)d_in[27];

  float* out = (float*)d_out;                 // [2,3,256,256] then depth [2,64,64]
  float* ws = (float*)d_ws;

  // workspace carve (floats)
  size_t off = 0;
  float* mlp_o = ws + off;  off += 2097152;   // [524288, 4]
  float* x0    = ws + off;  off += 24576;     // [2,3,64,64]
  float* xa    = ws + off;  off += 524288;    // [2,64,64,64]
  float* xskip = ws + off;  off += 524288;
  float* ytmp  = ws + off;  off += 524288;
  float* x128  = ws + off;  off += 2097152;   // [2,64,128,128]
  float* up2o  = ws + off;  off += 8388608;   // [2,256,128,128]
  float* x256  = ws + off;  off += 8388608;   // [2,64,256,256]
  float* up1o  = mlp_o;                       // reuse: [2,256,64,64] == 2097152

  // --- stage 1: MLP (WMMA) ---
  mlp_kernel<<<16384, 64, 0, stream>>>(rays, W_in, b_in, W_dense, b_dense,
                                       W_reg, b_reg, W_out, b_out, mlp_o);
  // --- stage 2: volume rendering ---
  render_kernel<<<32, 256, 0, stream>>>(mlp_o, t_vals, x0,
                                        out + (size_t)2 * 3 * 256 * 256);
  // --- stage 3: SRGAN generator (WMMA implicit-GEMM convs, TDM weight tiles) ---
  dim3 cblk(128);
  // conv1 9x9 3->64 + PReLU                (npix 4096 -> 64 pix tiles)
  conv_wmma_tiled<<<dim3(64, 1, 2), cblk, 0, stream>>>(
      x0, g_conv1_w, g_conv1_b, nullptr, g_prelu1, xa, 3, 64, 64, 64, 9);
  (void)hipMemcpyAsync(xskip, xa, sizeof(float) * 524288,
                       hipMemcpyDeviceToDevice, stream);
  // 4 residual blocks
  for (int k = 0; k < 4; ++k) {
    conv_wmma_tiled<<<dim3(64, 1, 2), cblk, 0, stream>>>(
        xa, res_wa + (size_t)k * 64 * 64 * 9, res_ba + k * 64,
        nullptr, res_alpha + k, ytmp, 64, 64, 64, 64, 3);
    conv_wmma_tiled<<<dim3(64, 1, 2), cblk, 0, stream>>>(
        ytmp, res_wb + (size_t)k * 64 * 64 * 9, res_bb + k * 64,
        xa, nullptr, xa, 64, 64, 64, 64, 3);
  }
  // mid conv + skip
  conv_wmma_tiled<<<dim3(64, 1, 2), cblk, 0, stream>>>(
      xa, g_mid_w, g_mid_b, xskip, nullptr, ytmp, 64, 64, 64, 64, 3);
  // up1: conv 64->256, shuffle x2 + PReLU
  conv_wmma_tiled<<<dim3(64, 4, 2), cblk, 0, stream>>>(
      ytmp, up1_w, up1_b, nullptr, nullptr, up1o, 64, 256, 64, 64, 3);
  shuffle_prelu_kernel<<<(2 * 64 * 128 * 128 + 255) / 256, 256, 0, stream>>>(
      up1o, up1_a, x128, 64, 64, 64);
  // up2: conv 64->256 at 128x128, shuffle x2 + PReLU
  conv_wmma_tiled<<<dim3(256, 4, 2), cblk, 0, stream>>>(
      x128, up2_w, up2_b, nullptr, nullptr, up2o, 64, 256, 128, 128, 3);
  shuffle_prelu_kernel<<<(2 * 64 * 256 * 256 + 255) / 256, 256, 0, stream>>>(
      up2o, up2_a, x256, 64, 128, 128);
  // final 9x9 64->3 conv straight into d_out
  conv_wmma_tiled<<<dim3(1024, 1, 2), cblk, 0, stream>>>(
      x256, g_out_w, g_out_b, nullptr, nullptr, out, 64, 3, 256, 256, 9);
}
